// GraphSAGE_LinkPredictor_21199958573768
// MI455X (gfx1250) — matrix operand order repaired and checked by reading, not compiled, not run on hardware
//
#include <hip/hip_runtime.h>
#include <hip/hip_bf16.h>
#include <math.h>

// GraphSAGE link predictor for gfx1250 (MI455X).
// Edge aggregation: f32 hardware atomics (L2-resident working set).
// Dense transforms: v_wmma_f32_16x16x32_bf16, f32 accumulate, with weight
// matrices pre-packed once per call into per-lane WMMA B-fragments.

#define N_NODES 100000
#define N_EDGES 1600000
#define N_PAIRS 200000
// IN_CH == HID == 64 everywhere.

typedef __attribute__((ext_vector_type(16))) __bf16 v16bf;
typedef __attribute__((ext_vector_type(8)))  float  v8f;

// ---------------- utility kernels ----------------

__global__ void zero_f32(float* __restrict__ p, int n) {
    int i = blockIdx.x * blockDim.x + threadIdx.x;
    if (i < n) p[i] = 0.0f;
}

__global__ void degree_count(const int* __restrict__ dst, float* __restrict__ cnt) {
    int e = blockIdx.x * blockDim.x + threadIdx.x;
    if (e < N_EDGES) unsafeAtomicAdd(&cnt[dst[e]], 1.0f);
}

// one thread per (edge, channel): coalesced 256B gather per edge,
// global_atomic_add_f32 (no return) scatter into agg.
__global__ void scatter_add(const float* __restrict__ feat,
                            const int* __restrict__ src,
                            const int* __restrict__ dst,
                            float* __restrict__ agg) {
    int t = blockIdx.x * blockDim.x + threadIdx.x;
    int e = t >> 6;
    int c = t & 63;
    if (e < N_EDGES) {
        float v = feat[src[e] * 64 + c];
        unsafeAtomicAdd(&agg[dst[e] * 64 + c], v);
    }
}

// ---------------- weight fragment pre-pack ----------------
// B fragment: 32x16 bf16 tile from row-major W[K][64] (ISA 7.12.2).
// Lane holds one column; lanes 0-15 cover K=kbase..+15 (b[j] = K=kbase+j),
// lanes 16-31 cover K=kbase+16..+31.
// Packed layout: fragment id = nt*kslices + ks; per lane 16 bf16 (32 B),
// contiguous across the wave -> consumer loads one b128 pair per lane.
__global__ void __launch_bounds__(32)
pack_b_frag(const float* __restrict__ W, v16bf* __restrict__ out, int kslices) {
    int lane = threadIdx.x;
    int nt = blockIdx.x / kslices;
    int ks = blockIdx.x % kslices;
    int kbase = ks * 32;
    int col  = nt * 16 + (lane & 15);
    int koff = (lane >> 4) * 16;
    const float* p = W + (size_t)(kbase + koff) * 64 + col;
    v16bf b;
#pragma unroll
    for (int j = 0; j < 16; ++j) b[j] = (__bf16)(p[j * 64]);
    out[(size_t)blockIdx.x * 32 + lane] = b;
}

// ---------------- A-fragment loader ----------------
// A fragment: 16x32 bf16 tile. Lane holds one row; khalf = (lane>>4)*16 selects
// K = khalf+{0..15} (a[j] = K=khalf+j). `p` = per-lane row pointer + kbase.
__device__ __forceinline__ v16bf load_a(const float* __restrict__ p, int lane, float scale) {
    int khalf = (lane >> 4) * 16;
    v16bf a;
#pragma unroll
    for (int j = 0; j < 8; ++j) a[j]     = (__bf16)(p[khalf + j] * scale);
#pragma unroll
    for (int j = 0; j < 8; ++j) a[8 + j] = (__bf16)(p[8 + khalf + j] * scale);
    return a;
}

// ---------------- SAGE dense transform ----------------
// out[row, :] = act( (agg[row,:]/max(cnt,1)) @ Wl + x[row,:] @ Wr + bias )
// 128 threads = 4 waves; block handles 16 rows; wave w handles out cols
// [16w, 16w+16). K=64 swept in two K=32 WMMA steps per operand pair.
template <int DO_RELU>
__global__ void __launch_bounds__(128)
sage_wmma(const float* __restrict__ x, const float* __restrict__ agg,
          const float* __restrict__ cnt,
          const v16bf* __restrict__ fragWl, const v16bf* __restrict__ fragWr,
          const float* __restrict__ bias, float* __restrict__ out) {
    int lane = threadIdx.x & 31;
    int wave = threadIdx.x >> 5;
    int row0 = blockIdx.x * 16;

    int arow  = row0 + (lane & 15);
    float deg = cnt[arow];
    float scl = 1.0f / fmaxf(deg, 1.0f);

    v8f acc = {};
#pragma unroll
    for (int ks = 0; ks < 2; ++ks) {
        int kbase = ks * 32;
        size_t fidx = (size_t)(wave * 2 + ks) * 32 + lane;
        v16bf am = load_a(agg + (size_t)arow * 64 + kbase, lane, scl);
        v16bf bl = fragWl[fidx];
        acc = __builtin_amdgcn_wmma_f32_16x16x32_bf16(false, am, false, bl,
                                                      (short)0, acc, false, false);
        v16bf ax = load_a(x + (size_t)arow * 64 + kbase, lane, 1.0f);
        v16bf br = fragWr[fidx];
        acc = __builtin_amdgcn_wmma_f32_16x16x32_bf16(false, ax, false, br,
                                                      (short)0, acc, false, false);
    }

    // C/D layout: lane holds col wave*16+(lane&15); VGPR r -> row (lane>>4)*8+r.
    int col   = wave * 16 + (lane & 15);
    int rbase = (lane >> 4) * 8;
    float bv  = bias[col];
#pragma unroll
    for (int r = 0; r < 8; ++r) {
        float v = acc[r] + bv;
        if (DO_RELU) v = fmaxf(v, 0.0f);
        out[(size_t)(row0 + rbase + r) * 64 + col] = v;
    }
}

// ---------------- pair MLP ----------------
// pair = [h[p0] | h[p1]] (K=128); z = relu(pair @ Wp1 + bp1) via WMMA;
// logit = z @ Wp2 + bp2 via per-pair dot through LDS; out = sigmoid(logit).
__global__ void __launch_bounds__(128)
pair_mlp(const float* __restrict__ h,
         const int* __restrict__ p0, const int* __restrict__ p1,
         const v16bf* __restrict__ fragWp1, const float* __restrict__ bp1,
         const float* __restrict__ Wp2, const float* __restrict__ bp2,
         float* __restrict__ out) {
    __shared__ float zbuf[16][64];

    int lane  = threadIdx.x & 31;
    int wave  = threadIdx.x >> 5;
    int prow0 = blockIdx.x * 16;

    int mypair = prow0 + (lane & 15);
    int r0 = p0[mypair];
    int r1 = p1[mypair];

    v8f acc = {};
#pragma unroll
    for (int ks = 0; ks < 4; ++ks) {
        int kbase = ks * 32;   // K slices 0-31,32-63 from h[p0]; 64-95,96-127 from h[p1]
        const float* arow = (ks < 2) ? (h + (size_t)r0 * 64 + kbase)
                                     : (h + (size_t)r1 * 64 + (kbase - 64));
        v16bf a = load_a(arow, lane, 1.0f);
        v16bf b = fragWp1[(size_t)(wave * 4 + ks) * 32 + lane];
        acc = __builtin_amdgcn_wmma_f32_16x16x32_bf16(false, a, false, b,
                                                      (short)0, acc, false, false);
    }

    int col   = wave * 16 + (lane & 15);
    int rbase = (lane >> 4) * 8;
    float bv  = bp1[col];
#pragma unroll
    for (int r = 0; r < 8; ++r)
        zbuf[rbase + r][col] = fmaxf(acc[r] + bv, 0.0f);

    __syncthreads();

    int t = threadIdx.x;
    if (t < 16) {
        float s = bp2[0];
#pragma unroll 8
        for (int k = 0; k < 64; ++k) s += zbuf[t][k] * Wp2[k];
        out[prow0 + t] = 1.0f / (1.0f + __expf(-s));
    }
}

// ---------------- launch ----------------

extern "C" void kernel_launch(void* const* d_in, const int* in_sizes, int n_in,
                              void* d_out, int out_size, void* d_ws, size_t ws_size,
                              hipStream_t stream) {
    const float* x         = (const float*)d_in[0];
    const int*   edge_idx  = (const int*)d_in[1];
    const int*   edge_pair = (const int*)d_in[2];
    const float* W1l = (const float*)d_in[3];
    const float* W1r = (const float*)d_in[4];
    const float* b1  = (const float*)d_in[5];
    const float* W2l = (const float*)d_in[6];
    const float* W2r = (const float*)d_in[7];
    const float* b2  = (const float*)d_in[8];
    const float* Wp1 = (const float*)d_in[9];
    const float* bp1 = (const float*)d_in[10];
    const float* Wp2 = (const float*)d_in[11];
    const float* bp2 = (const float*)d_in[12];

    const int* src = edge_idx;             // edge_index[0]
    const int* dst = edge_idx + N_EDGES;   // edge_index[1]
    const int* p0  = edge_pair;            // edge_pairs[0]
    const int* p1  = edge_pair + N_PAIRS;  // edge_pairs[1]

    // workspace layout (floats): agg[N*64] | cnt[N] | h1[N*64] | h2[N*64]
    // then packed bf16 B-fragments (32B-aligned: 77,200,000 % 32 == 0).
    float* agg = (float*)d_ws;
    float* cnt = agg + (size_t)N_NODES * 64;
    float* h1  = cnt + N_NODES;
    float* h2  = h1 + (size_t)N_NODES * 64;
    v16bf* fw1l = (v16bf*)(h2 + (size_t)N_NODES * 64);
    v16bf* fw1r = fw1l + 8 * 32;    // 4 ntiles * 2 kslices
    v16bf* fw2l = fw1r + 8 * 32;
    v16bf* fw2r = fw2l + 8 * 32;
    v16bf* fwp1 = fw2r + 8 * 32;    // 4 ntiles * 4 kslices
    float* outf = (float*)d_out;

    // pre-pack weights into WMMA B-fragments (bf16)
    pack_b_frag<<<8, 32, 0, stream>>>(W1l, fw1l, 2);
    pack_b_frag<<<8, 32, 0, stream>>>(W1r, fw1r, 2);
    pack_b_frag<<<8, 32, 0, stream>>>(W2l, fw2l, 2);
    pack_b_frag<<<8, 32, 0, stream>>>(W2r, fw2r, 2);
    pack_b_frag<<<16, 32, 0, stream>>>(Wp1, fwp1, 4);

    // zero agg + cnt (contiguous), build degree counts (shared by both layers)
    int nz = N_NODES * 65;
    zero_f32<<<(nz + 255) / 256, 256, 0, stream>>>(agg, nz);
    degree_count<<<(N_EDGES + 255) / 256, 256, 0, stream>>>(dst, cnt);

    // layer 1
    scatter_add<<<(N_EDGES * 64) / 256, 256, 0, stream>>>(x, src, dst, agg);
    sage_wmma<1><<<N_NODES / 16, 128, 0, stream>>>(x, agg, cnt, fw1l, fw1r, b1, h1);

    // layer 2
    zero_f32<<<(N_NODES * 64 + 255) / 256, 256, 0, stream>>>(agg, N_NODES * 64);
    scatter_add<<<(N_EDGES * 64) / 256, 256, 0, stream>>>(h1, src, dst, agg);
    sage_wmma<0><<<N_NODES / 16, 128, 0, stream>>>(h1, agg, cnt, fw2l, fw2r, b2, h2);

    // link predictor
    pair_mlp<<<N_PAIRS / 16, 128, 0, stream>>>(h2, p0, p1, fwp1, bp1, Wp2, bp2, outf);
}
